// cost_model_55722905698634
// MI455X (gfx1250) — compile-verified, matrix operand order checked by reading, not run on hardware
//
#include <hip/hip_runtime.h>
#include <hip/hip_bf16.h>

// ---------------------------------------------------------------------------
// CDNA5 (gfx1250) wave32 WMMA types
// ---------------------------------------------------------------------------
typedef __attribute__((ext_vector_type(16))) __bf16 v16bf;
typedef __attribute__((ext_vector_type(8)))  float  v8f;
typedef unsigned short ushort_t;

#define N_NODES 4096
#define N_EDGES 131072
#define IN_FEATS 128
#define HIDDEN 256
#define HEADS 8
#define DH 32
#define FF 2048
#define LN_EPS 1e-5f

// f32 pair -> packed 2x bf16 (RNE), merged via v_perm_b32
__device__ __forceinline__ unsigned pack2bf(float a, float b) {
  unsigned x = __float_as_uint(a);
  unsigned y = __float_as_uint(b);
  x += 0x7FFFu + ((x >> 16) & 1u);
  y += 0x7FFFu + ((y >> 16) & 1u);
  return __builtin_amdgcn_perm(y, x, 0x07060302u);  // [x.hi16 | y.hi16]
}
__device__ __forceinline__ ushort_t bf16u(float a) {
  unsigned x = __float_as_uint(a);
  x += 0x7FFFu + ((x >> 16) & 1u);
  return (ushort_t)(x >> 16);
}

// Load one WMMA bf16 operand row-fragment (16 bf16 per lane) from a bf16 row.
// Lane pattern (ISA 7.12.2): lanes 0-15 hold K {0..7,16..23}, lanes 16-31 hold
// K {8..15,24..31}; caller passes klo = kbase + half*8. Two b128 loads, no VALU.
__device__ __forceinline__ v16bf load_op_bf16(const ushort_t* __restrict__ row, int klo) {
  union { uint4 q[2]; v16bf v; } r;
  r.q[0] = *(const uint4*)(row + klo);       // K klo .. klo+7
  r.q[1] = *(const uint4*)(row + klo + 16);  // K klo+16 .. klo+23
  return r.v;
}

__device__ __forceinline__ v8f wmma_bf16(v16bf a, v16bf b, v8f c) {
  return __builtin_amdgcn_wmma_f32_16x16x32_bf16(false, a, false, b, (short)0, c,
                                                 false, false);
}

__device__ __forceinline__ float rmax16(float v) {
  v = fmaxf(v, __shfl_xor(v, 1));
  v = fmaxf(v, __shfl_xor(v, 2));
  v = fmaxf(v, __shfl_xor(v, 4));
  v = fmaxf(v, __shfl_xor(v, 8));
  return v;
}
__device__ __forceinline__ float rsum16(float v) {
  v += __shfl_xor(v, 1);
  v += __shfl_xor(v, 2);
  v += __shfl_xor(v, 4);
  v += __shfl_xor(v, 8);
  return v;
}
__device__ __forceinline__ float rsum32(float v) {
  v += __shfl_xor(v, 1);
  v += __shfl_xor(v, 2);
  v += __shfl_xor(v, 4);
  v += __shfl_xor(v, 8);
  v += __shfl_xor(v, 16);
  return v;
}

// ---------------------------------------------------------------------------
// WMMA GEMM: D[M,N] = A[M,K] @ W[N,K]^T (+ A2@W2^T) + bias (+resid f32), relu?
// bf16 inputs; one 16x64 output strip per wave (4 WMMA tiles share A fragment).
// Optional f32 and/or bf16 outputs. N%64==0, K%32==0, M%16==0.
// ---------------------------------------------------------------------------
__global__ void gemm_wmma_kernel(const ushort_t* __restrict__ A,
                                 const ushort_t* __restrict__ W,
                                 const float* __restrict__ bias,
                                 const ushort_t* __restrict__ A2,
                                 const ushort_t* __restrict__ W2,
                                 const float* __restrict__ resid,
                                 float* __restrict__ Df,
                                 ushort_t* __restrict__ Db,
                                 int M, int N, int K, int relu) {
  int wave = (int)((blockIdx.x * blockDim.x + threadIdx.x) >> 5);
  int tilesN = N >> 6;
  int total = (M >> 4) * tilesN;
  if (wave >= total) return;
  int tm = wave / tilesN, tn = wave % tilesN;
  int lane = threadIdx.x & 31, hf = lane >> 4, r = lane & 15;

  v8f c[4];
#pragma unroll
  for (int j = 0; j < 4; ++j) c[j] = (v8f){0.f, 0.f, 0.f, 0.f, 0.f, 0.f, 0.f, 0.f};

  {
    const ushort_t* aRow = A + (size_t)(tm * 16 + r) * K;
    const ushort_t* wRow0 = W + (size_t)(tn * 64 + r) * K;
    for (int kk = 0; kk < K; kk += 32) {
      __builtin_prefetch(aRow + kk + 32, 0, 1);
      v16bf a = load_op_bf16(aRow, kk + hf * 8);
#pragma unroll
      for (int j = 0; j < 4; ++j) {
        v16bf b = load_op_bf16(wRow0 + (size_t)j * 16 * K, kk + hf * 8);
        c[j] = wmma_bf16(a, b, c[j]);
      }
    }
  }
  if (A2 != nullptr) {
    const ushort_t* aRow = A2 + (size_t)(tm * 16 + r) * K;
    const ushort_t* wRow0 = W2 + (size_t)(tn * 64 + r) * K;
    for (int kk = 0; kk < K; kk += 32) {
      __builtin_prefetch(aRow + kk + 32, 0, 1);
      v16bf a = load_op_bf16(aRow, kk + hf * 8);
#pragma unroll
      for (int j = 0; j < 4; ++j) {
        v16bf b = load_op_bf16(wRow0 + (size_t)j * 16 * K, kk + hf * 8);
        c[j] = wmma_bf16(a, b, c[j]);
      }
    }
  }
  // C/D layout: lane -> n = (lane%16); VGPR v -> m = v + 8*half
#pragma unroll
  for (int j = 0; j < 4; ++j) {
    int n = tn * 64 + j * 16 + r;
    float bv = bias ? bias[n] : 0.f;
#pragma unroll
    for (int v = 0; v < 8; ++v) {
      int m = tm * 16 + v + 8 * hf;
      float val = c[j][v] + bv;
      if (resid) val += resid[(size_t)m * N + n];
      if (relu) val = fmaxf(val, 0.f);
      if (Df) Df[(size_t)m * N + n] = val;
      if (Db) Db[(size_t)m * N + n] = bf16u(val);
    }
  }
}

// ---------------------------------------------------------------------------
// Flash attention over bf16 QKV (row = [q|k|v], ld=768 bf16), bf16 ctx out.
// One (head, 16-query tile) per wave.
// ---------------------------------------------------------------------------
__global__ void attn_flash_kernel(const ushort_t* __restrict__ QKV,
                                  ushort_t* __restrict__ CTX, int N) {
  __shared__ float sP[8][16 * 32];
  int wid = threadIdx.x >> 5;
  int wave = blockIdx.x * 8 + wid;
  int nq = N >> 4;
  int h = wave / nq;
  int qt = wave % nq;
  int lane = threadIdx.x & 31, hf = lane >> 4, r = lane & 15;
  const float scale = 0.17677669529663687f;  // 1/sqrt(32)
  const size_t ld = 3 * HIDDEN;

  const ushort_t* qRow = QKV + (size_t)(qt * 16 + r) * ld + h * DH;
  v16bf aq = load_op_bf16(qRow, hf * 8);

  v8f c0 = {0.f, 0.f, 0.f, 0.f, 0.f, 0.f, 0.f, 0.f};
  v8f c1 = {0.f, 0.f, 0.f, 0.f, 0.f, 0.f, 0.f, 0.f};
  float mrun[8], lrun[8];
#pragma unroll
  for (int v = 0; v < 8; ++v) { mrun[v] = -1e30f; lrun[v] = 0.f; }
  float* sw = sP[wid];

  for (int kb = 0; kb < N; kb += 32) {
    const ushort_t* kRow0 = QKV + (size_t)(kb + r) * ld + HIDDEN + h * DH;
    const ushort_t* kRow1 = QKV + (size_t)(kb + 16 + r) * ld + HIDDEN + h * DH;
    v16bf b0 = load_op_bf16(kRow0, hf * 8);
    v16bf b1 = load_op_bf16(kRow1, hf * 8);
    v8f z = {0.f, 0.f, 0.f, 0.f, 0.f, 0.f, 0.f, 0.f};
    v8f s0 = wmma_bf16(aq, b0, z);
    v8f s1 = wmma_bf16(aq, b1, z);

    // online softmax per row (row m lives on fixed VGPR across a 16-lane half)
#pragma unroll
    for (int v = 0; v < 8; ++v) {
      float x0 = s0[v] * scale, x1 = s1[v] * scale;
      float t = rmax16(fmaxf(x0, x1));
      float mnew = fmaxf(mrun[v], t);
      float al = __expf(mrun[v] - mnew);
      float p0 = __expf(x0 - mnew);
      float p1 = __expf(x1 - mnew);
      float rs = rsum16(p0 + p1);
      lrun[v] = lrun[v] * al + rs;
      mrun[v] = mnew;
      c0[v] *= al;
      c1[v] *= al;
      sw[(v + 8 * hf) * 32 + r] = p0;
      sw[(v + 8 * hf) * 32 + 16 + r] = p1;
    }
    __syncthreads();

    // Re-layout P (16x32) from LDS into an A operand (f32 -> bf16 pack)
    union { unsigned u[8]; v16bf v; } ap;
    const float* pr = sw + r * 32 + hf * 8;
#pragma unroll
    for (int j = 0; j < 4; ++j) ap.u[j] = pack2bf(pr[2 * j], pr[2 * j + 1]);
#pragma unroll
    for (int j = 0; j < 4; ++j) ap.u[4 + j] = pack2bf(pr[16 + 2 * j], pr[17 + 2 * j]);

    // V chunk as B operands (lane = output dim column, K dim = 32 keys)
    const ushort_t* vb = QKV + (size_t)kb * ld + 2 * HIDDEN + h * DH;
    union { unsigned u[8]; v16bf v; } bv0, bv1;
#pragma unroll
    for (int j = 0; j < 8; ++j) {
      int kA = (j < 4) ? (hf * 8 + 2 * j) : (16 + hf * 8 + 2 * (j - 4));
      unsigned lo0 = vb[(size_t)kA * ld + r];
      unsigned hi0 = vb[(size_t)(kA + 1) * ld + r];
      unsigned lo1 = vb[(size_t)kA * ld + 16 + r];
      unsigned hi1 = vb[(size_t)(kA + 1) * ld + 16 + r];
      bv0.u[j] = lo0 | (hi0 << 16);
      bv1.u[j] = lo1 | (hi1 << 16);
    }
    c0 = wmma_bf16(ap.v, bv0.v, c0);
    c1 = wmma_bf16(ap.v, bv1.v, c1);
    __syncthreads();
  }

#pragma unroll
  for (int v = 0; v < 8; ++v) {
    int m = qt * 16 + v + 8 * hf;
    float inv = 1.f / lrun[v];
    CTX[(size_t)m * HIDDEN + h * DH + r] = bf16u(c0[v] * inv);
    CTX[(size_t)m * HIDDEN + h * DH + 16 + r] = bf16u(c1[v] * inv);
  }
}

// ---------------------------------------------------------------------------
// Small helper kernels
// ---------------------------------------------------------------------------
__global__ void fill0_kernel(float* __restrict__ p, long long n) {
  long long i = (long long)blockIdx.x * blockDim.x + threadIdx.x;
  if (i < n) p[i] = 0.f;
}

// f32 -> packed bf16 pairs
__global__ void cvt_bf16_kernel(const float* __restrict__ in,
                                unsigned* __restrict__ out, long long n2) {
  long long i = (long long)blockIdx.x * blockDim.x + threadIdx.x;
  if (i < n2) {
    float2 f = ((const float2*)in)[i];
    out[i] = pack2bf(f.x, f.y);
  }
}

__global__ void deg_kernel(const int* __restrict__ dst, float* __restrict__ deg, int E) {
  int e = blockIdx.x * blockDim.x + threadIdx.x;
  if (e < E) atomicAdd(&deg[dst[e]], 1.f);
}

__global__ void scatter_add_kernel(const float* __restrict__ H,
                                   const int* __restrict__ src,
                                   const int* __restrict__ dst,
                                   float* __restrict__ agg, int D, int E) {
  long long t = (long long)blockIdx.x * blockDim.x + threadIdx.x;
  if (t >= (long long)E * D) return;
  int e = (int)(t / D);
  int d = (int)(t % D);
  atomicAdd(&agg[(size_t)dst[e] * D + d], H[(size_t)src[e] * D + d]);
}

// h_neigh = agg/deg, emitted directly as packed bf16 (D even)
__global__ void divdeg_bf_kernel(const float* __restrict__ agg,
                                 const float* __restrict__ deg,
                                 unsigned* __restrict__ hnb, int D, long long n2) {
  long long i = (long long)blockIdx.x * blockDim.x + threadIdx.x;
  if (i >= n2) return;
  long long idx = 2 * i;
  float dg = fmaxf(deg[idx / D], 1.f);
  float a = agg[idx] / dg;
  float b = agg[idx + 1] / dg;
  hnb[i] = pack2bf(a, b);
}

// in-place LayerNorm (f32), one wave per 256-wide row, + bf16 mirror store
__global__ void layernorm_kernel(float* __restrict__ X,
                                 const float* __restrict__ g,
                                 const float* __restrict__ b,
                                 ushort_t* __restrict__ Xb, int M) {
  int wid = threadIdx.x >> 5;
  int row = blockIdx.x * 8 + wid;
  if (row >= M) return;
  int lane = threadIdx.x & 31;
  float x[8];
  float s = 0.f;
#pragma unroll
  for (int j = 0; j < 8; ++j) {
    x[j] = X[(size_t)row * HIDDEN + j * 32 + lane];
    s += x[j];
  }
  float mu = rsum32(s) * (1.f / HIDDEN);
  float v = 0.f;
#pragma unroll
  for (int j = 0; j < 8; ++j) {
    float d = x[j] - mu;
    v += d * d;
  }
  float rstd = rsqrtf(rsum32(v) * (1.f / HIDDEN) + LN_EPS);
#pragma unroll
  for (int j = 0; j < 8; ++j) {
    int col = j * 32 + lane;
    float val = g[col] * (x[j] - mu) * rstd + b[col];
    X[(size_t)row * HIDDEN + col] = val;
    if (Xb) Xb[(size_t)row * HIDDEN + col] = bf16u(val);
  }
}

// partial column sums: each block covers 256 rows, atomic accumulate
__global__ void colsum_kernel(const float* __restrict__ X, float* __restrict__ out,
                              int M, int D) {
  int c = threadIdx.x;
  int r0 = blockIdx.x * 256;
  float s = 0.f;
  for (int m = r0; m < r0 + 256 && m < M; ++m) s += X[(size_t)m * D + c];
  atomicAdd(&out[c], s);
}

__global__ void final_out_kernel(const float* __restrict__ hsum,
                                 const float* __restrict__ W,
                                 const float* __restrict__ b,
                                 float* __restrict__ out) {
  int lane = threadIdx.x & 31;
  float s = 0.f;
  for (int i = lane; i < HIDDEN; i += 32) s += hsum[i] * W[i];
  s = rsum32(s);
  if (lane == 0) out[0] = fmaxf(s + b[0], 0.f);
}

// ---------------------------------------------------------------------------
// Host orchestration
// ---------------------------------------------------------------------------
static inline void launch_gemm(const ushort_t* A, const ushort_t* W, const float* bias,
                               const ushort_t* A2, const ushort_t* W2,
                               const float* resid, float* Df, ushort_t* Db,
                               int M, int N, int K, int relu, hipStream_t stream) {
  int tiles = (M / 16) * (N / 64);
  int blocks = (tiles + 7) / 8;
  gemm_wmma_kernel<<<blocks, 256, 0, stream>>>(A, W, bias, A2, W2, resid, Df, Db,
                                               M, N, K, relu);
}

static inline void launch_cvt(const float* src, ushort_t* dst, long long n,
                              hipStream_t stream) {
  long long n2 = n / 2;
  cvt_bf16_kernel<<<(int)((n2 + 255) / 256), 256, 0, stream>>>(src, (unsigned*)dst, n2);
}

extern "C" void kernel_launch(void* const* d_in, const int* in_sizes, int n_in,
                              void* d_out, int out_size, void* d_ws, size_t ws_size,
                              hipStream_t stream) {
  (void)in_sizes; (void)n_in; (void)out_size; (void)ws_size;
  const float* xin = (const float*)d_in[0];
  const int* src = (const int*)d_in[1];
  const int* dst = (const int*)d_in[2];
  int idx = 3;
  const float *sageWs[3], *sageWn[3], *sageB[3];
  for (int l = 0; l < 3; ++l) {
    sageWs[l] = (const float*)d_in[idx++];
    sageWn[l] = (const float*)d_in[idx++];
    sageB[l]  = (const float*)d_in[idx++];
  }
  const float *ffW[3], *ffB[3];
  for (int l = 0; l < 3; ++l) {
    ffW[l] = (const float*)d_in[idx++];
    ffB[l] = (const float*)d_in[idx++];
  }
  const float *Wqkv[2], *bqkv[2], *Wo[2], *bo[2], *Wf1[2], *bf1[2], *Wf2[2], *bf2[2];
  const float *g1[2], *be1[2], *g2[2], *be2[2];
  for (int t = 0; t < 2; ++t) {
    Wqkv[t] = (const float*)d_in[idx++]; bqkv[t] = (const float*)d_in[idx++];
    Wo[t] = (const float*)d_in[idx++];   bo[t] = (const float*)d_in[idx++];
    Wf1[t] = (const float*)d_in[idx++];  bf1[t] = (const float*)d_in[idx++];
    Wf2[t] = (const float*)d_in[idx++];  bf2[t] = (const float*)d_in[idx++];
    g1[t] = (const float*)d_in[idx++];   be1[t] = (const float*)d_in[idx++];
    g2[t] = (const float*)d_in[idx++];   be2[t] = (const float*)d_in[idx++];
  }
  const float* outW = (const float*)d_in[idx++];
  const float* outB = (const float*)d_in[idx++];

  // ---- workspace layout ----
  const size_t NH = (size_t)N_NODES * HIDDEN;       // 1,048,576
  float* X = (float*)d_ws;
  float* Y = X + NH;
  float* AGG = Y + NH;
  float* DEG = AGG + NH;
  float* HSUM = DEG + N_NODES;
  ushort_t* bp = (ushort_t*)(HSUM + HIDDEN);
  ushort_t* Xb   = bp; bp += NH;
  ushort_t* Yb   = bp; bp += NH;
  ushort_t* HNb  = bp; bp += NH;
  ushort_t* CTXb = bp; bp += NH;
  ushort_t* QKVb = bp; bp += (size_t)N_NODES * 3 * HIDDEN;
  ushort_t* MIDb = bp; bp += (size_t)N_NODES * FF;
  ushort_t* XINb = bp; bp += (size_t)N_NODES * IN_FEATS;
  ushort_t *WsB[3], *WnB[3], *ffWB[3], *WqkvB[2], *WoB[2], *Wf1B[2], *Wf2B[2];
  for (int l = 0; l < 3; ++l) {
    size_t k = (l == 0) ? IN_FEATS : HIDDEN;
    WsB[l] = bp; bp += HIDDEN * k;
    WnB[l] = bp; bp += HIDDEN * k;
  }
  for (int l = 0; l < 3; ++l) { ffWB[l] = bp; bp += (size_t)HIDDEN * HIDDEN; }
  for (int t = 0; t < 2; ++t) {
    WqkvB[t] = bp; bp += (size_t)3 * HIDDEN * HIDDEN;
    WoB[t] = bp;   bp += (size_t)HIDDEN * HIDDEN;
    Wf1B[t] = bp;  bp += (size_t)FF * HIDDEN;
    Wf2B[t] = bp;  bp += (size_t)HIDDEN * FF;
  }

  const int E = N_EDGES;

  // ---- one-shot bf16 mirrors of weights + input features ----
  for (int l = 0; l < 3; ++l) {
    long long k = (l == 0) ? IN_FEATS : HIDDEN;
    launch_cvt(sageWs[l], WsB[l], (long long)HIDDEN * k, stream);
    launch_cvt(sageWn[l], WnB[l], (long long)HIDDEN * k, stream);
  }
  for (int l = 0; l < 3; ++l) launch_cvt(ffW[l], ffWB[l], (long long)HIDDEN * HIDDEN, stream);
  for (int t = 0; t < 2; ++t) {
    launch_cvt(Wqkv[t], WqkvB[t], (long long)3 * HIDDEN * HIDDEN, stream);
    launch_cvt(Wo[t], WoB[t], (long long)HIDDEN * HIDDEN, stream);
    launch_cvt(Wf1[t], Wf1B[t], (long long)FF * HIDDEN, stream);
    launch_cvt(Wf2[t], Wf2B[t], (long long)HIDDEN * FF, stream);
  }
  launch_cvt(xin, XINb, (long long)N_NODES * IN_FEATS, stream);

  // degree (shared by all SAGE layers)
  fill0_kernel<<<(N_NODES + 255) / 256, 256, 0, stream>>>(DEG, N_NODES);
  deg_kernel<<<(E + 255) / 256, 256, 0, stream>>>(dst, DEG, E);

  // ---- SAGE 0: input(128) -> X (+Xb) ----
  {
    long long n = (long long)N_NODES * IN_FEATS;
    fill0_kernel<<<(int)((n + 255) / 256), 256, 0, stream>>>(AGG, n);
    long long ed = (long long)E * IN_FEATS;
    scatter_add_kernel<<<(int)((ed + 255) / 256), 256, 0, stream>>>(xin, src, dst, AGG, IN_FEATS, E);
    divdeg_bf_kernel<<<(int)((n / 2 + 255) / 256), 256, 0, stream>>>(AGG, DEG, (unsigned*)HNb, IN_FEATS, n / 2);
    launch_gemm(XINb, WsB[0], sageB[0], HNb, WnB[0], nullptr, X, Xb,
                N_NODES, HIDDEN, IN_FEATS, 0, stream);
  }
  // ---- SAGE 1: X -> Y (+Yb) ----
  {
    long long n = (long long)NH;
    fill0_kernel<<<(int)((n + 255) / 256), 256, 0, stream>>>(AGG, n);
    long long ed = (long long)E * HIDDEN;
    scatter_add_kernel<<<(int)((ed + 255) / 256), 256, 0, stream>>>(X, src, dst, AGG, HIDDEN, E);
    divdeg_bf_kernel<<<(int)((n / 2 + 255) / 256), 256, 0, stream>>>(AGG, DEG, (unsigned*)HNb, HIDDEN, n / 2);
    launch_gemm(Xb, WsB[1], sageB[1], HNb, WnB[1], nullptr, Y, Yb,
                N_NODES, HIDDEN, HIDDEN, 0, stream);
  }
  // ---- SAGE 2: Y -> X (+Xb) ----
  {
    long long n = (long long)NH;
    fill0_kernel<<<(int)((n + 255) / 256), 256, 0, stream>>>(AGG, n);
    long long ed = (long long)E * HIDDEN;
    scatter_add_kernel<<<(int)((ed + 255) / 256), 256, 0, stream>>>(Y, src, dst, AGG, HIDDEN, E);
    divdeg_bf_kernel<<<(int)((n / 2 + 255) / 256), 256, 0, stream>>>(AGG, DEG, (unsigned*)HNb, HIDDEN, n / 2);
    launch_gemm(Yb, WsB[2], sageB[2], HNb, WnB[2], nullptr, X, Xb,
                N_NODES, HIDDEN, HIDDEN, 0, stream);
  }

  // ---- FF relu layers: X -> Y -> X -> Y ----
  launch_gemm(Xb, ffWB[0], ffB[0], nullptr, nullptr, nullptr, Y, Yb, N_NODES, HIDDEN, HIDDEN, 1, stream);
  launch_gemm(Yb, ffWB[1], ffB[1], nullptr, nullptr, nullptr, X, Xb, N_NODES, HIDDEN, HIDDEN, 1, stream);
  launch_gemm(Xb, ffWB[2], ffB[2], nullptr, nullptr, nullptr, Y, Yb, N_NODES, HIDDEN, HIDDEN, 1, stream);

  // ---- Transformer encoder layers: (Y,Yb) -> (Y,Yb) ----
  for (int t = 0; t < 2; ++t) {
    // QKV: bf16-only output (only ever a WMMA input)
    launch_gemm(Yb, WqkvB[t], bqkv[t], nullptr, nullptr, nullptr, nullptr, QKVb,
                N_NODES, 3 * HIDDEN, HIDDEN, 0, stream);
    attn_flash_kernel<<<(HEADS * (N_NODES / 16)) / 8, 256, 0, stream>>>(QKVb, CTXb, N_NODES);
    // attn out + residual -> X (f32), then LN in place (+Xb mirror)
    launch_gemm(CTXb, WoB[t], bo[t], nullptr, nullptr, Y, X, nullptr,
                N_NODES, HIDDEN, HIDDEN, 0, stream);
    layernorm_kernel<<<N_NODES / 8, 256, 0, stream>>>(X, g1[t], be1[t], Xb, N_NODES);
    // FFN: mid bf16-only; second GEMM fuses residual, then LN (+Yb mirror)
    launch_gemm(Xb, Wf1B[t], bf1[t], nullptr, nullptr, nullptr, nullptr, MIDb,
                N_NODES, FF, HIDDEN, 1, stream);
    launch_gemm(MIDb, Wf2B[t], bf2[t], nullptr, nullptr, X, Y, nullptr,
                N_NODES, HIDDEN, FF, 0, stream);
    layernorm_kernel<<<N_NODES / 8, 256, 0, stream>>>(Y, g2[t], be2[t], Yb, N_NODES);
  }

  // ---- reduce + output head ----
  fill0_kernel<<<1, 256, 0, stream>>>(HSUM, HIDDEN);
  colsum_kernel<<<N_NODES / 256, 256, 0, stream>>>(Y, HSUM, N_NODES, HIDDEN);
  final_out_kernel<<<1, 32, 0, stream>>>(HSUM, outW, outB, (float*)d_out);
}
